// THAN_35244501631055
// MI455X (gfx1250) — matrix operand order repaired and checked by reading, not compiled, not run on hardware
//
#include <hip/hip_runtime.h>
#include <math.h>

// ---------------------------------------------------------------------------
// THAN forward for MI455X (gfx1250, wave32).
// All dense GEMMs run on v_wmma_f32_16x16x32_f16. Weight matrices (B) are
// pre-packed once per GEMM into an f16 transposed layout so each B fragment
// is one contiguous v16h load (2x global_load_b128); each wave computes NT
// adjacent 16x16 tiles so the A fragment (f32 loads + cvt_pk_f16) is
// amortized over NT WMMAs. Irregular graph ops (edge softmax, segment sums,
// gathers) are VALU + global atomics — no matrix structure there.
// THANLayer GATs have dst == arange(N): single-edge softmax => alpha == 1
// exactly in f32, so tl_Wdst / tl_al / tl_ar are algebraically dead.
// Workspace requirement: ~380 MB of float scratch in d_ws.
// ---------------------------------------------------------------------------

typedef _Float16 v16h __attribute__((ext_vector_type(16)));
typedef float    v8f  __attribute__((ext_vector_type(8)));

constexpr int kN = 20000, kNPID = 10000, kNOD = 10000, kNO = 2000, kND = 2000;
constexpr int kEM = 160000, kEODG = 40000;
constexpr int kIN = 128, kH = 8, kD = 64, kHD = 512, kOUT = 16;

static inline int cdiv(int a, int b) { return (a + b - 1) / b; }

// ---------------------------------------------------------------------------
// B pack: Bt[n*K + k] = (f16) B[k*Nc + n]   (K % 32 == 0 -> rows 64B aligned)
// ---------------------------------------------------------------------------
__global__ void k_bt16(const float* __restrict__ B, _Float16* __restrict__ Bt,
                       int K, int Nc) {
  int i = blockIdx.x * blockDim.x + threadIdx.x;
  if (i >= K * Nc) return;
  int n = i / K, k = i - n * K;
  Bt[i] = (_Float16)B[(size_t)k * Nc + n];
}

// ---------------------------------------------------------------------------
// WMMA GEMM: C[M,Nc] = A[M,K] @ B[K,Nc] (+bias), B pre-packed f16 transposed.
// One wave computes NT adjacent 16x16 tiles (one A fragment, NT B fragments,
// NT v_wmma per K-step of 32).
// Fragment layouts used here (wave32; A/B use contiguous k order, not the ISA table's; C/D per ISA):
//   A 16x32 f16: m = lane&15; half idx: K = (lane>>4)*16 + idx
//   B 32x16 f16: n = lane&15; half idx: K = (lane>>4)*16 + idx  -> contiguous in Bt
//   C/D 16x16 f32: vgpr r: row = r + (lane>>4)*8, col = lane&15
// ---------------------------------------------------------------------------
template <int NT>
__global__ void k_gemm_wmma(const float* __restrict__ A, const _Float16* __restrict__ Bt,
                            const float* __restrict__ bias, float* __restrict__ C,
                            int M, int K, int Nc) {
  int wid  = (blockIdx.x * blockDim.x + threadIdx.x) >> 5;
  int lane = threadIdx.x & 31;
  int groupsN = (Nc >> 4) / NT;
  int tiles   = (M >> 4) * groupsN;
  if (wid >= tiles) return;                 // wave-uniform: EXEC stays all-1s
  int m0 = (wid / groupsN) << 4;
  int n0 = (wid % groupsN) * (16 * NT);
  int lh  = lane >> 4;                      // half-wave select (0/1)
  int l15 = lane & 15;
  int mrow = m0 + l15;
  v8f acc[NT];
#pragma unroll
  for (int t = 0; t < NT; ++t) acc[t] = (v8f){};
  for (int k0 = 0; k0 < K; k0 += 32) {
    // A fragment: 4x global_load_b128 + 8x v_cvt_pk_f16_f32, shared by NT wmmas
    const float* arow = A + (size_t)mrow * K + k0 + lh * 16;
    float4 a0 = *(const float4*)(arow + 0);
    float4 a1 = *(const float4*)(arow + 4);
    float4 a2 = *(const float4*)(arow + 8);
    float4 a3 = *(const float4*)(arow + 12);
    v16h af;
    af[0]  = (_Float16)a0.x; af[1]  = (_Float16)a0.y; af[2]  = (_Float16)a0.z; af[3]  = (_Float16)a0.w;
    af[4]  = (_Float16)a1.x; af[5]  = (_Float16)a1.y; af[6]  = (_Float16)a1.z; af[7]  = (_Float16)a1.w;
    af[8]  = (_Float16)a2.x; af[9]  = (_Float16)a2.y; af[10] = (_Float16)a2.z; af[11] = (_Float16)a2.w;
    af[12] = (_Float16)a3.x; af[13] = (_Float16)a3.y; af[14] = (_Float16)a3.z; af[15] = (_Float16)a3.w;
#pragma unroll
    for (int t = 0; t < NT; ++t) {
      // B fragment: 16 contiguous halfs -> 2x global_load_b128, no converts
      v16h bf = *(const v16h*)(Bt + (size_t)(n0 + t * 16 + l15) * K + k0 + lh * 16);
      acc[t] = __builtin_amdgcn_wmma_f32_16x16x32_f16(false, af, false, bf,
                                                      (short)0, acc[t], false, false);
    }
  }
#pragma unroll
  for (int t = 0; t < NT; ++t) {
    int nc = n0 + t * 16 + l15;
    float bv = bias ? bias[nc] : 0.f;
#pragma unroll
    for (int r = 0; r < 8; ++r)
      C[(size_t)(m0 + r + lh * 8) * Nc + nc] = acc[t][r] + bv;
  }
}

// ---------------------------------------------------------------------------
// Elementwise / graph kernels
// ---------------------------------------------------------------------------
__global__ void k_fill(float* p, float v, int n) {
  int i = blockIdx.x * blockDim.x + threadIdx.x;
  if (i < n) p[i] = v;
}

__device__ __forceinline__ void atomicMaxF(float* addr, float v) {
  if (v >= 0.f) atomicMax((int*)addr, __float_as_int(v));
  else          atomicMin((unsigned int*)addr, __float_as_uint(v));
}

// out[n,h] = sum_d f[n,h,d] * a[h,d]
__global__ void k_score(const float* __restrict__ f, const float* __restrict__ a,
                        float* __restrict__ out, int nh) {
  int t = blockIdx.x * blockDim.x + threadIdx.x;
  if (t >= nh) return;
  int n = t >> 3, h = t & 7;
  const float* fr = f + (size_t)n * kHD + h * kD;
  const float* ar = a + h * kD;
  float s = 0.f;
#pragma unroll 8
  for (int d = 0; d < kD; ++d) s += fr[d] * ar[d];
  out[t] = s;
}

// edge score = leaky_relu(es[src]+ed[dst], 0.2); running per-dst max
__global__ void k_edge_max(const int* __restrict__ src, const int* __restrict__ dst,
                           const float* __restrict__ es, const float* __restrict__ ed,
                           float* __restrict__ eb, float* __restrict__ m, int E) {
  int e = blockIdx.x * blockDim.x + threadIdx.x;
  if (e >= E) return;
  int s = src[e], d = dst[e];
#pragma unroll
  for (int h = 0; h < kH; ++h) {
    float v = es[s * kH + h] + ed[d * kH + h];
    v = v > 0.f ? v : 0.2f * v;
    eb[(size_t)e * kH + h] = v;
    atomicMaxF(&m[d * kH + h], v);
  }
}

__global__ void k_edge_sum(const int* __restrict__ dst, const float* __restrict__ eb,
                           const float* __restrict__ m, float* __restrict__ sb, int E) {
  int e = blockIdx.x * blockDim.x + threadIdx.x;
  if (e >= E) return;
  int d = dst[e];
#pragma unroll
  for (int h = 0; h < kH; ++h) {
    float mm = m[d * kH + h];
    mm = mm < -3e38f ? 0.f : mm;             // isfinite(m) ? m : 0
    atomicAdd(&sb[d * kH + h], expf(eb[(size_t)e * kH + h] - mm));
  }
}

// out[n,j] = b[j] (+ residual[n,j])
__global__ void k_init_out(float* out, const float* b, const float* res, int n512) {
  int i = blockIdx.x * blockDim.x + threadIdx.x;
  if (i >= n512) return;
  out[i] = b[i & (kHD - 1)] + (res ? res[i] : 0.f);
}

// one wave per edge: out[dst, :512] += fs[src, :512] * alpha[h]
__global__ void k_aggregate(const int* __restrict__ src, const int* __restrict__ dst,
                            const float* __restrict__ eb, const float* __restrict__ m,
                            const float* __restrict__ sb, const float* __restrict__ fs,
                            float* __restrict__ out, int E) {
  int w = (blockIdx.x * blockDim.x + threadIdx.x) >> 5;
  int lane = threadIdx.x & 31;
  if (w >= E) return;
  int s = src[w], d = dst[w];
  float alpha[kH];
#pragma unroll
  for (int h = 0; h < kH; ++h) {
    float mm = m[d * kH + h];
    mm = mm < -3e38f ? 0.f : mm;
    alpha[h] = expf(eb[(size_t)w * kH + h] - mm) / (sb[d * kH + h] + 1e-9f);
  }
  const float* fr = fs + (size_t)s * kHD;
  float* orow = out + (size_t)d * kHD;
#pragma unroll
  for (int i = 0; i < 16; ++i) {
    int j = lane + 32 * i;
    atomicAdd(&orow[j], fr[j] * alpha[j >> 6]);
  }
}

__global__ void k_act(float* p, int n, int mode) {   // 0 = elu, 1 = lrelu(0.01)
  int i = blockIdx.x * blockDim.x + threadIdx.x;
  if (i >= n) return;
  float x = p[i];
  p[i] = (mode == 0) ? (x > 0.f ? x : expm1f(x)) : (x > 0.f ? x : 0.01f * x);
}

__global__ void k_head_mean(const float* __restrict__ in, float* __restrict__ out, int nd) {
  int i = blockIdx.x * blockDim.x + threadIdx.x;
  if (i >= nd) return;
  int n = i >> 6, d = i & 63;
  float s = 0.f;
#pragma unroll
  for (int h = 0; h < kH; ++h) s += in[(size_t)n * kHD + h * kD + d];
  out[i] = s * 0.125f;
}

// w += (1/N) * sum_k tanh(t1[n,k]) * W2[k]
__global__ void k_sem_reduce(const float* __restrict__ t1, const float* __restrict__ W2,
                             float* wacc, int n, int Kd, float invN) {
  int r = blockIdx.x * blockDim.x + threadIdx.x;
  if (r >= n) return;
  float s = 0.f;
  for (int k = 0; k < Kd; ++k) s += tanhf(t1[(size_t)r * Kd + k]) * W2[k];
  atomicAdd(wacc, s * invN);
}

__global__ void k_softmax2(float* w) {
  if (blockIdx.x | threadIdx.x) return;
  float a = w[0], b = w[1];
  float mx = fmaxf(a, b);
  float ea = expf(a - mx), eb2 = expf(b - mx);
  float inv = 1.f / (ea + eb2);
  w[0] = ea * inv; w[1] = eb2 * inv;
}

__global__ void k_combine2(const float* __restrict__ z0, const float* __restrict__ z1,
                           const float* __restrict__ beta, float* __restrict__ out, int n) {
  int i = blockIdx.x * blockDim.x + threadIdx.x;
  if (i >= n) return;
  out[i] = beta[0] * z0[i] + beta[1] * z1[i];
}

__global__ void k_deg(const int* __restrict__ idx, float* deg, int E) {
  int e = blockIdx.x * blockDim.x + threadIdx.x;
  if (e < E) atomicAdd(&deg[idx[e]], 1.f);
}

__global__ void k_rsqrt_scale(const float* __restrict__ x, const float* __restrict__ deg,
                              float* __restrict__ y, int nd) {
  int i = blockIdx.x * blockDim.x + threadIdx.x;
  if (i >= nd) return;
  y[i] = x[i] * rsqrtf(fmaxf(deg[i >> 6], 1.f));
}

__global__ void k_edge_gather64(const int* __restrict__ src, const int* __restrict__ dst,
                                const float* __restrict__ xh, float* __restrict__ agg, int E) {
  int w = (blockIdx.x * blockDim.x + threadIdx.x) >> 5;
  int lane = threadIdx.x & 31;
  if (w >= E) return;
  int s = src[w], d = dst[w];
  atomicAdd(&agg[(size_t)d * 64 + lane],      xh[(size_t)s * 64 + lane]);
  atomicAdd(&agg[(size_t)d * 64 + 32 + lane], xh[(size_t)s * 64 + 32 + lane]);
}

__global__ void k_gather_cat(const int* __restrict__ oi, const int* __restrict__ di,
                             const float* __restrict__ o2, const float* __restrict__ d2,
                             float* __restrict__ cat, int nd) {
  int i = blockIdx.x * blockDim.x + threadIdx.x;
  if (i >= nd) return;
  int n = i >> 6, d = i & 63;
  cat[(size_t)n * 128 + d]      = o2[(size_t)oi[n] * 64 + d];
  cat[(size_t)n * 128 + 64 + d] = d2[(size_t)di[n] * 64 + d];
}

// THANLayer GAT with single-edge softmax: alpha == 1 exactly.
// e[n,j] = lrelu(fs[map[n],j] + resh[n,j] + b[j], 0.01)
__global__ void k_tl_combine(const int* __restrict__ map, const float* __restrict__ fs,
                             const float* __restrict__ resh, const float* __restrict__ b,
                             float* __restrict__ e, int n512) {
  int i = blockIdx.x * blockDim.x + threadIdx.x;
  if (i >= n512) return;
  int n = i >> 9, j = i & 511;
  float v = fs[(size_t)map[n] * kHD + j] + resh[i] + b[j];
  e[i] = v > 0.f ? v : 0.01f * v;
}

__global__ void k_concat2(const float* __restrict__ a, const float* __restrict__ b,
                          float* __restrict__ cat, int n1024) {
  int i = blockIdx.x * blockDim.x + threadIdx.x;
  if (i >= n1024) return;
  int n = i >> 10, j = i & 1023;
  cat[i] = (j < kHD) ? a[(size_t)n * kHD + j] : b[(size_t)n * kHD + j - kHD];
}

// BN(eval, mean0/var1) -> lrelu(0.01) -> softmax over 16, in place
__global__ void k_final(float* out, const float* __restrict__ gamma,
                        const float* __restrict__ beta, int n) {
  int r = blockIdx.x * blockDim.x + threadIdx.x;
  if (r >= n) return;
  const float bns = 0.9999950000374997f;   // 1/sqrt(1+1e-5)
  float* row = out + (size_t)r * kOUT;
  float v[kOUT]; float mx = -3.4e38f;
#pragma unroll
  for (int j = 0; j < kOUT; ++j) {
    float x = row[j] * gamma[j] * bns + beta[j];
    x = x > 0.f ? x : 0.01f * x;
    v[j] = x; mx = fmaxf(mx, x);
  }
  float ss = 0.f;
#pragma unroll
  for (int j = 0; j < kOUT; ++j) { v[j] = expf(v[j] - mx); ss += v[j]; }
  float inv = 1.f / ss;
#pragma unroll
  for (int j = 0; j < kOUT; ++j) row[j] = v[j] * inv;
}

// ---------------------------------------------------------------------------
// Host-side helpers
// ---------------------------------------------------------------------------
static void gemm(const float* A, const float* B, const float* bias, float* C,
                 int M, int K, int Nc, _Float16* bt, hipStream_t s) {
  k_bt16<<<cdiv(K * Nc, 256), 256, 0, s>>>(B, bt, K, Nc);
  int tilesN = Nc / 16;
  if (tilesN % 4 == 0) {
    int tiles = (M / 16) * (tilesN / 4);
    k_gemm_wmma<4><<<cdiv(tiles * 32, 128), 128, 0, s>>>(A, bt, bias, C, M, K, Nc);
  } else if (tilesN % 2 == 0) {
    int tiles = (M / 16) * (tilesN / 2);
    k_gemm_wmma<2><<<cdiv(tiles * 32, 128), 128, 0, s>>>(A, bt, bias, C, M, K, Nc);
  } else {
    int tiles = (M / 16) * tilesN;
    k_gemm_wmma<1><<<cdiv(tiles * 32, 128), 128, 0, s>>>(A, bt, bias, C, M, K, Nc);
  }
}

struct GatBufs { float *es, *ed, *mb, *sb, *eb; };

static void gat_attention(const int* src, const int* dst, int E,
                          const float* fs, const float* fd, int n_src, int n_dst,
                          const float* al, const float* ar,
                          const float* bvec, const float* res,
                          float* out, int actmode, GatBufs g, hipStream_t st) {
  k_score<<<cdiv(n_src * kH, 256), 256, 0, st>>>(fs, al, g.es, n_src * kH);
  k_score<<<cdiv(n_dst * kH, 256), 256, 0, st>>>(fd, ar, g.ed, n_dst * kH);
  k_fill<<<cdiv(n_dst * kH, 256), 256, 0, st>>>(g.mb, -INFINITY, n_dst * kH);
  k_fill<<<cdiv(n_dst * kH, 256), 256, 0, st>>>(g.sb, 0.f, n_dst * kH);
  k_edge_max<<<cdiv(E, 256), 256, 0, st>>>(src, dst, g.es, g.ed, g.eb, g.mb, E);
  k_edge_sum<<<cdiv(E, 256), 256, 0, st>>>(dst, g.eb, g.mb, g.sb, E);
  k_init_out<<<cdiv(n_dst * kHD, 256), 256, 0, st>>>(out, bvec, res, n_dst * kHD);
  k_aggregate<<<cdiv(E * 32, 256), 256, 0, st>>>(src, dst, g.eb, g.mb, g.sb, fs, out, E);
  k_act<<<cdiv(n_dst * kHD, 256), 256, 0, st>>>(out, n_dst * kHD, actmode);
}

extern "C" void kernel_launch(void* const* d_in, const int* in_sizes, int n_in,
                              void* d_out, int out_size, void* d_ws, size_t ws_size,
                              hipStream_t stream) {
  (void)in_sizes; (void)n_in; (void)out_size; (void)ws_size;
  const float* h        = (const float*)d_in[0];
  const float* pid_h    = (const float*)d_in[1];
  const float* o_h      = (const float*)d_in[2];
  const float* d_h      = (const float*)d_in[3];
  // d_in[4] od_h_in: unused by the reference forward
  const float* mpgat_W  = (const float*)d_in[5];
  const float* mpgat_al = (const float*)d_in[6];
  const float* mpgat_ar = (const float*)d_in[7];
  const float* mpgat_b  = (const float*)d_in[8];
  const float* sem_W1   = (const float*)d_in[9];
  const float* sem_b1   = (const float*)d_in[10];
  const float* sem_W2   = (const float*)d_in[11];
  const float* gc_od_W  = (const float*)d_in[12];
  const float* gc_od_b  = (const float*)d_in[13];
  const float* gc_do_W  = (const float*)d_in[14];
  const float* gc_do_b  = (const float*)d_in[15];
  const float* rec_o_W  = (const float*)d_in[16];
  const float* rec_o_b  = (const float*)d_in[17];
  const float* rec_d_W  = (const float*)d_in[18];
  const float* rec_d_b  = (const float*)d_in[19];
  const float* rec_odcat_W = (const float*)d_in[20];
  const float* rec_odcat_b = (const float*)d_in[21];
  const float* bp_Wsrc  = (const float*)d_in[22];
  const float* bp_Wdst  = (const float*)d_in[23];
  const float* bp_al    = (const float*)d_in[24];
  const float* bp_ar    = (const float*)d_in[25];
  const float* bp_res   = (const float*)d_in[26];
  const float* bp_b     = (const float*)d_in[27];
  const float* rec_pid_W = (const float*)d_in[28];
  const float* rec_pid_b = (const float*)d_in[29];
  const float* rec_od_W  = (const float*)d_in[30];
  const float* rec_od_b  = (const float*)d_in[31];
  const float* tl_Wsrc  = (const float*)d_in[32];
  // d_in[33] tl_Wdst, d_in[34] tl_al, d_in[35] tl_ar: dead (single-edge softmax)
  const float* tl_res   = (const float*)d_in[36];
  const float* tl_b     = (const float*)d_in[37];
  const float* nt_W1    = (const float*)d_in[38];
  const float* nt_b1    = (const float*)d_in[39];
  const float* nt_W2    = (const float*)d_in[40];
  const float* pred1_W  = (const float*)d_in[41];
  const float* pred1_b  = (const float*)d_in[42];
  const float* bn_gamma = (const float*)d_in[43];
  const float* bn_beta  = (const float*)d_in[44];
  const int* mp_src[2]  = { (const int*)d_in[45], (const int*)d_in[47] };
  const int* mp_dst[2]  = { (const int*)d_in[46], (const int*)d_in[48] };
  const int* od_src     = (const int*)d_in[49];
  const int* od_dst     = (const int*)d_in[50];
  const int* pa_dst     = (const int*)d_in[51];
  const int* pf_dst     = (const int*)d_in[52];
  const int* od_o_idx   = (const int*)d_in[53];
  const int* od_d_idx   = (const int*)d_in[54];
  float* out = (float*)d_out;
  float* ws  = (float*)d_ws;

  // --------------------------- workspace layout (floats) -------------------
  constexpr size_t NS = (size_t)kN * kHD;                 // 10,240,000
  float* emb0 = ws;                                       // [N,512]
  float* emb1 = ws + NS;                                  // [N,512]
  float* h1   = ws + 2 * NS;                              // [N,512]
  float* rA   = ws + 3 * NS;   // fs scratch for HAN; later e0
  float* rB   = ws + 4 * NS;   // resh scratch;       later m_h
  float* rC   = ws + 5 * NS;   // e1
  float* cat  = ws + 6 * NS;                              // [N,1024]
  float* tbuf = ws + 8 * NS;                              // [N,128] sem scratch
  float* sp   = tbuf + (size_t)kN * kIN;                  // small pool
  float* es   = sp;                  float* ed  = sp + 160000;
  float* mb   = sp + 320000;         float* sb  = sp + 480000;
  float* eb   = sp + 640000;                              // [160000,8]
  float* degs = sp + 1920000;        float* degd = sp + 1922000;
  float* xh   = sp + 1924000;        float* agg  = sp + 2052000;
  float* res0 = sp + 2180000;        float* res1 = sp + 2308000;
  float* oh2  = sp + 2436000;        float* dh2  = sp + 2564000;
  float* odf  = sp + 2692000;                             // [NOD,64]
  float* r01  = sp + 3332000;                             // [10000,64]
  float* pidh2 = sp + 3972000;       float* odh2 = sp + 4612000;
  float* fstl = sp + 5252000;                             // [10000,512]
  float* wacc = sp + 10372000;                            // [2]
  _Float16* bt = (_Float16*)(sp + 10372032);              // f16 B^T pack, 64K halfs
  // bipartite-GAT buffers alias emb0/emb1 (dead after h1 is formed)
  float* fs_bp  = emb0;              float* fd_bp  = emb0 + (size_t)kNPID * kHD;
  float* res_bp = emb1;              float* out_bp = emb1 + (size_t)kNPID * kHD;
  GatBufs gb = { es, ed, mb, sb, eb };

  // ---------------- Phase 1: HAN layer (2 metapath GATs) -------------------
  float* emb[2] = { emb0, emb1 };
  for (int i = 0; i < 2; ++i) {
    gemm(h, mpgat_W + (size_t)i * kIN * kHD, nullptr, rA, kN, kIN, kHD, bt, stream);
    gat_attention(mp_src[i], mp_dst[i], kEM, rA, rA, kN, kN,
                  mpgat_al + i * kHD, mpgat_ar + i * kHD,
                  mpgat_b + i * kHD, nullptr, emb[i], /*elu*/0, gb, stream);
  }
  // semantic attention -> h1
  k_fill<<<1, 32, 0, stream>>>(wacc, 0.f, 2);
  for (int i = 0; i < 2; ++i) {
    gemm(emb[i], sem_W1, sem_b1, tbuf, kN, kHD, 128, bt, stream);
    k_sem_reduce<<<cdiv(kN, 256), 256, 0, stream>>>(tbuf, sem_W2, wacc + i, kN, 128, 1.f / kN);
  }
  k_softmax2<<<1, 32, 0, stream>>>(wacc);
  k_combine2<<<cdiv((int)NS, 256), 256, 0, stream>>>(emb0, emb1, wacc, h1, (int)NS);

  // ---------------- Phase 2: o<->d GraphConvs + recover --------------------
  k_fill<<<cdiv(kNO, 256), 256, 0, stream>>>(degs, 0.f, kNO);
  k_fill<<<cdiv(kND, 256), 256, 0, stream>>>(degd, 0.f, kND);
  k_deg<<<cdiv(kEODG, 256), 256, 0, stream>>>(od_src, degs, kEODG);
  k_deg<<<cdiv(kEODG, 256), 256, 0, stream>>>(od_dst, degd, kEODG);
  // res0 = gconv(o_h, od_src->od_dst)
  k_rsqrt_scale<<<cdiv(kNO * 64, 256), 256, 0, stream>>>(o_h, degs, xh, kNO * 64);
  k_fill<<<cdiv(kND * 64, 256), 256, 0, stream>>>(agg, 0.f, kND * 64);
  k_edge_gather64<<<cdiv(kEODG * 32, 256), 256, 0, stream>>>(od_src, od_dst, xh, agg, kEODG);
  k_rsqrt_scale<<<cdiv(kND * 64, 256), 256, 0, stream>>>(agg, degd, agg, kND * 64);
  gemm(agg, gc_od_W, gc_od_b, res0, kND, 64, 64, bt, stream);
  // res1 = gconv(d_h, od_dst->od_src)
  k_rsqrt_scale<<<cdiv(kND * 64, 256), 256, 0, stream>>>(d_h, degd, xh, kND * 64);
  k_fill<<<cdiv(kNO * 64, 256), 256, 0, stream>>>(agg, 0.f, kNO * 64);
  k_edge_gather64<<<cdiv(kEODG * 32, 256), 256, 0, stream>>>(od_dst, od_src, xh, agg, kEODG);
  k_rsqrt_scale<<<cdiv(kNO * 64, 256), 256, 0, stream>>>(agg, degs, agg, kNO * 64);
  gemm(agg, gc_do_W, gc_do_b, res1, kNO, 64, 64, bt, stream);
  gemm(res1, rec_o_W, rec_o_b, oh2, kNO, 64, 64, bt, stream);
  gemm(res0, rec_d_W, rec_d_b, dh2, kND, 64, 64, bt, stream);
  // od_feat = concat(o_h2[oi], d_h2[di]) @ rec_odcat_W + b
  k_gather_cat<<<cdiv(kNOD * 64, 256), 256, 0, stream>>>(od_o_idx, od_d_idx, oh2, dh2, tbuf, kNOD * 64);
  gemm(tbuf, rec_odcat_W, rec_odcat_b, odf, kNOD, 128, 64, bt, stream);

  // ---------------- Phase 3: bipartite GATs od<->pid -----------------------
  // r0: src=od_feat (pf_dst), dst=pid (pa_dst)
  gemm(odf,   bp_Wsrc, nullptr, fs_bp,  kNOD,  64, kHD, bt, stream);
  gemm(pid_h, bp_Wdst, nullptr, fd_bp,  kNPID, 64, kHD, bt, stream);
  gemm(pid_h, bp_res,  nullptr, res_bp, kNPID, 64, kHD, bt, stream);
  gat_attention(pf_dst, pa_dst, kN, fs_bp, fd_bp, kNOD, kNPID,
                bp_al, bp_ar, bp_b, res_bp, out_bp, /*lrelu*/1, gb, stream);
  k_head_mean<<<cdiv(kNPID * 64, 256), 256, 0, stream>>>(out_bp, r01, kNPID * 64);
  gemm(r01, rec_pid_W, rec_pid_b, pidh2, kNPID, 64, 64, bt, stream);
  // r1: src=pid (pa_dst), dst=od_feat (pf_dst)
  constexpr size_t WB = (size_t)64 * kHD;
  gemm(pid_h, bp_Wsrc + WB, nullptr, fs_bp,  kNPID, 64, kHD, bt, stream);
  gemm(odf,   bp_Wdst + WB, nullptr, fd_bp,  kNOD,  64, kHD, bt, stream);
  gemm(odf,   bp_res  + WB, nullptr, res_bp, kNOD,  64, kHD, bt, stream);
  gat_attention(pa_dst, pf_dst, kN, fs_bp, fd_bp, kNPID, kNOD,
                bp_al + kHD, bp_ar + kHD, bp_b + kHD, res_bp, out_bp, 1, gb, stream);
  k_head_mean<<<cdiv(kNOD * 64, 256), 256, 0, stream>>>(out_bp, r01, kNOD * 64);
  gemm(r01, rec_od_W, rec_od_b, odh2, kNOD, 64, 64, bt, stream);

  // ---------------- Phase 4: THANLayer (alpha==1 simplification) -----------
  gemm(pidh2, tl_Wsrc, nullptr, fstl, kNPID, 64, kHD, bt, stream);
  gemm(h, tl_res, nullptr, rB, kN, kIN, kHD, bt, stream);
  k_tl_combine<<<cdiv((int)NS, 256), 256, 0, stream>>>(pa_dst, fstl, rB, tl_b, rA, (int)NS);
  gemm(odh2, tl_Wsrc + WB, nullptr, fstl, kNOD, 64, kHD, bt, stream);
  gemm(h, tl_res + (size_t)kIN * kHD, nullptr, rB, kN, kIN, kHD, bt, stream);
  k_tl_combine<<<cdiv((int)NS, 256), 256, 0, stream>>>(pf_dst, fstl, rB, tl_b + kHD, rC, (int)NS);
  // nodetype semantic attention -> m_h (in rB)
  k_fill<<<1, 32, 0, stream>>>(wacc, 0.f, 2);
  gemm(rA, nt_W1, nt_b1, tbuf, kN, kHD, 64, bt, stream);
  k_sem_reduce<<<cdiv(kN, 256), 256, 0, stream>>>(tbuf, nt_W2, wacc + 0, kN, 64, 1.f / kN);
  gemm(rC, nt_W1, nt_b1, tbuf, kN, kHD, 64, bt, stream);
  k_sem_reduce<<<cdiv(kN, 256), 256, 0, stream>>>(tbuf, nt_W2, wacc + 1, kN, 64, 1.f / kN);
  k_softmax2<<<1, 32, 0, stream>>>(wacc);
  k_combine2<<<cdiv((int)NS, 256), 256, 0, stream>>>(rA, rC, wacc, rB, (int)NS);

  // ---------------- Phase 5: prediction head -------------------------------
  k_concat2<<<cdiv(kN * 1024, 256), 256, 0, stream>>>(rB, h1, cat, kN * 1024);
  gemm(cat, pred1_W, pred1_b, out, kN, 1024, kOUT, bt, stream);
  k_final<<<cdiv(kN, 256), 256, 0, stream>>>(out, bn_gamma, bn_beta, kN);
}
